// MambaBlock_18098992185577
// MI455X (gfx1250) — compile-verified, matrix operand order checked by reading, not run on hardware
//
#include <hip/hip_runtime.h>
#include <hip/hip_bf16.h>

// ---------------------------------------------------------------------------
// MambaBlock forward for MI455X (gfx1250, wave32, WMMA).
//   H=512, D=S=1024, K=4, B=8, L=256  ->  BL = 2048 rows
// GEMMs use v_wmma_f32_16x16x32_bf16 with all operands pre-staged as bf16:
//   - A operands stored row-major bf16 -> per-lane K-stripe = 2x b128 loads
//   - B operands stored TRANSPOSED ([N][K]) bf16 -> same 2x b128 pattern
// Each wave computes a 16x64 C strip (4 accumulators, A reused 4x).
// The [L,D,S] A_bar tensor (1 GB) is never materialized; exp() is fused
// into the einsum with xc staged into LDS via gfx1250 async global->LDS
// loads (ASYNCcnt) when the toolchain exposes the builtins.
// ---------------------------------------------------------------------------

#define Hc 512
#define Dc 1024
#define Sc 1024
#define Bc 8
#define Lc 256
#define BLc 2048
#define EPSc 1e-3f

typedef __attribute__((ext_vector_type(16))) __bf16 bf16x16;
typedef __attribute__((ext_vector_type(8)))  __bf16 bf16x8;
typedef __attribute__((ext_vector_type(8)))  float  f32x8;
typedef __attribute__((ext_vector_type(4)))  int    i32x4;

#define AS1 __attribute__((address_space(1)))
#define AS3 __attribute__((address_space(3)))

#if __has_builtin(__builtin_amdgcn_global_load_async_to_lds_b128) && \
    __has_builtin(__builtin_amdgcn_s_wait_asynccnt)
#define USE_ASYNC_LDS 1
#else
#define USE_ASYNC_LDS 0
#endif

static __device__ __forceinline__ float silu_f(float v) {
  return v * (1.0f / (1.0f + __expf(-v)));
}

// Per-lane WMMA 16-bit fragment (ISA 16x32 A layout; B uses the transposed
// copy so the same loader applies): base p = row/col pointer + kb.
static __device__ __forceinline__ bf16x16 load_frag16(
    const __bf16* __restrict__ p, int khalf8) {
  bf16x8 lo = *(const bf16x8*)(p + khalf8);
  bf16x8 hi = *(const bf16x8*)(p + 16 + khalf8);
  return __builtin_shufflevector(lo, hi, 0, 1, 2, 3, 4, 5, 6, 7,
                                 8, 9, 10, 11, 12, 13, 14, 15);
}

static __device__ __forceinline__ bf16x16 load_frag16_masked(
    const __bf16* __restrict__ p, int khalf8, bool valid) {
  bf16x8 z = {};
  bf16x8 lo = valid ? *(const bf16x8*)(p + khalf8) : z;
  bf16x8 hi = valid ? *(const bf16x8*)(p + 16 + khalf8) : z;
  return __builtin_shufflevector(lo, hi, 0, 1, 2, 3, 4, 5, 6, 7,
                                 8, 9, 10, 11, 12, 13, 14, 15);
}

// ---------------------------------------------------------------------------
// Kernel 0: tiled transpose + f32->bf16 convert.  src [T][R][C] -> dst [T][C][R]
// grid: (C/32, R/32, T), block 256 (32x8).
// ---------------------------------------------------------------------------
__global__ __launch_bounds__(256) void mamba_transpose_bf16(
    const float* __restrict__ src, __bf16* __restrict__ dst, int R, int C) {
  __shared__ float tile[32][33];
  const float* s = src + (size_t)blockIdx.z * R * C;
  __bf16* d = dst + (size_t)blockIdx.z * R * C;
  const int tx = threadIdx.x & 31;
  const int ty = threadIdx.x >> 5;        // 0..7
  const int cb = blockIdx.x * 32;
  const int rb = blockIdx.y * 32;
#pragma unroll
  for (int i = ty; i < 32; i += 8)
    tile[i][tx] = s[(size_t)(rb + i) * C + cb + tx];
  __syncthreads();
#pragma unroll
  for (int i = ty; i < 32; i += 8)
    d[(size_t)(cb + i) * R + rb + tx] = (__bf16)tile[tx][i];
}

// ---------------------------------------------------------------------------
// Kernel 1: LayerNorm over last axis (H=512); writes bf16 (GEMM-A operand).
// ---------------------------------------------------------------------------
__global__ __launch_bounds__(256) void mamba_layernorm(
    const float* __restrict__ x, const float* __restrict__ gamma,
    const float* __restrict__ beta, __bf16* __restrict__ xn) {
  const int row = blockIdx.x;
  const int tid = threadIdx.x;
  const float* xr = x + (size_t)row * Hc;

  float v0 = xr[tid];
  float v1 = xr[tid + 256];

  __shared__ float red[256];
  red[tid] = v0 + v1;
  __syncthreads();
  for (int s = 128; s > 0; s >>= 1) {
    if (tid < s) red[tid] += red[tid + s];
    __syncthreads();
  }
  const float mean = red[0] * (1.0f / Hc);
  __syncthreads();
  float d0 = v0 - mean, d1 = v1 - mean;
  red[tid] = d0 * d0 + d1 * d1;
  __syncthreads();
  for (int s = 128; s > 0; s >>= 1) {
    if (tid < s) red[tid] += red[tid + s];
    __syncthreads();
  }
  const float rstd = rsqrtf(red[0] * (1.0f / Hc) + EPSc);

  xn[(size_t)row * Hc + tid] =
      (__bf16)(d0 * rstd * gamma[tid] + beta[tid]);
  xn[(size_t)row * Hc + tid + 256] =
      (__bf16)(d1 * rstd * gamma[tid + 256] + beta[tid + 256]);
}

// ---------------------------------------------------------------------------
// Kernel 2: in_proj.  h = xn @ W_in + b_in -> split xh(bf16) / gate(f32).
// Wave computes 16x64 strip; K = H = 512.
// ---------------------------------------------------------------------------
__global__ __launch_bounds__(256) void mamba_inproj(
    const __bf16* __restrict__ xn, const __bf16* __restrict__ WinT,
    const float* __restrict__ b_in, __bf16* __restrict__ xh,
    float* __restrict__ gate) {
  const int lane = threadIdx.x & 31;
  const int wave = threadIdx.x >> 5;
  const int id   = blockIdx.x * 8 + wave;
  const int NT   = (2 * Dc) / 64;          // 32 strips per row-tile
  const int tm   = id / NT;                // 0..127
  const int n0   = (id % NT) * 64;

  const int row    = tm * 16 + (lane & 15);
  const int ncol   = lane & 15;
  const int khalf8 = (lane >> 4) * 8;
  const __bf16* arow = xn + (size_t)row * Hc;
  const __bf16* brow[4];
#pragma unroll
  for (int j = 0; j < 4; ++j)
    brow[j] = WinT + (size_t)(n0 + j * 16 + ncol) * Hc;

  f32x8 acc[4] = {};
  for (int kb = 0; kb < Hc; kb += 32) {
    __builtin_prefetch(arow + kb + 64, 0, 1);
    bf16x16 a = load_frag16(arow + kb, khalf8);
#pragma unroll
    for (int j = 0; j < 4; ++j) {
      bf16x16 b = load_frag16(brow[j] + kb, khalf8);
      acc[j] = __builtin_amdgcn_wmma_f32_16x16x32_bf16(
          false, a, false, b, (short)0, acc[j], false, false);
    }
  }

  const int mbase = tm * 16 + ((lane >> 4) ? 8 : 0);
  const bool isGate = (n0 >= Dc);          // wave-uniform (64 | Dc)
#pragma unroll
  for (int j = 0; j < 4; ++j) {
    const int n = n0 + j * 16 + ncol;
    const float bias = b_in[n];
    if (!isGate) {
#pragma unroll
      for (int i = 0; i < 8; ++i)
        xh[(size_t)(mbase + i) * Dc + n] = (__bf16)(acc[j][i] + bias);
    } else {
#pragma unroll
      for (int i = 0; i < 8; ++i)
        gate[(size_t)(mbase + i) * Dc + (n - Dc)] = acc[j][i] + bias;
    }
  }
}

// ---------------------------------------------------------------------------
// Kernel 3: causal dense conv1d (K=4) + SiLU as 4 shifted GEMMs.
//   xc[b,l,o] = silu( sum_k sum_i xh[b, l+k-3, i] * conv_w[k,i,o] )
// convT is per-tap transposed: convT[k][o][i].  Masked A rows for l+k-3<0.
// ---------------------------------------------------------------------------
__global__ __launch_bounds__(256) void mamba_conv_silu(
    const __bf16* __restrict__ xh, const __bf16* __restrict__ convT,
    float* __restrict__ xc) {
  const int lane = threadIdx.x & 31;
  const int wave = threadIdx.x >> 5;
  const int id   = blockIdx.x * 8 + wave;
  const int NT   = Dc / 64;                // 16
  const int tm   = id / NT;                // 0..127
  const int n0   = (id % NT) * 64;

  const int row    = tm * 16 + (lane & 15);
  const int l      = row & (Lc - 1);
  const int bbase  = row & ~(Lc - 1);
  const int ncol   = lane & 15;
  const int khalf8 = (lane >> 4) * 8;

  f32x8 acc[4] = {};
#pragma unroll
  for (int tap = 0; tap < 4; ++tap) {
    const int srcl   = l + tap - 3;
    const bool valid = (srcl >= 0);
    const __bf16* arow = xh + (size_t)(bbase + srcl) * Dc;
    const __bf16* brow[4];
#pragma unroll
    for (int j = 0; j < 4; ++j)
      brow[j] = convT + (size_t)tap * Dc * Dc + (size_t)(n0 + j * 16 + ncol) * Dc;
    for (int kb = 0; kb < Dc; kb += 32) {
      bf16x16 a = load_frag16_masked(arow + kb, khalf8, valid);
#pragma unroll
      for (int j = 0; j < 4; ++j) {
        bf16x16 b = load_frag16(brow[j] + kb, khalf8);
        acc[j] = __builtin_amdgcn_wmma_f32_16x16x32_bf16(
            false, a, false, b, (short)0, acc[j], false, false);
      }
    }
  }

  const int mbase = tm * 16 + ((lane >> 4) ? 8 : 0);
#pragma unroll
  for (int j = 0; j < 4; ++j) {
    const int n = n0 + j * 16 + ncol;
#pragma unroll
    for (int i = 0; i < 8; ++i)
      xc[(size_t)(mbase + i) * Dc + n] = silu_f(acc[j][i]);
  }
}

// ---------------------------------------------------------------------------
// Kernel 4a: Asp = softplus(A_param)
// ---------------------------------------------------------------------------
__global__ __launch_bounds__(256) void mamba_softplus(
    const float* __restrict__ A_param, float* __restrict__ Asp) {
  const int i = blockIdx.x * 256 + threadIdx.x;
  if (i < Dc * Sc) Asp[i] = log1pf(__expf(A_param[i]));
}

// ---------------------------------------------------------------------------
// Kernel 4b: pseudo-SSM einsum + gate, A_bar regenerated inline:
//   yg[b,n,s] = ( sum_d xc[b,n,d] * exp(Asp[d,s]*(n+1)) ) * silu(gate[b,n,s])
// Thread = (n, s); 8 batch accumulators; one v_exp_f32 feeds 8 FMAs.
// xc is staged into LDS with gfx1250 async global->LDS b128 loads
// (ASYNCcnt path) when available; VGPR round-trip otherwise.
// ---------------------------------------------------------------------------
__global__ __launch_bounds__(256) void mamba_ssm_gate(
    const float* __restrict__ xc, const float* __restrict__ Asp,
    const float* __restrict__ gate, __bf16* __restrict__ yg) {
  const int tid = threadIdx.x;
  const int n   = blockIdx.y;
  const int s   = blockIdx.x * 256 + tid;
  const float pos = (float)(n + 1);

  __shared__ float xcs[Bc][256];
  float acc[Bc] = {0.f, 0.f, 0.f, 0.f, 0.f, 0.f, 0.f, 0.f};

  for (int d0 = 0; d0 < Dc; d0 += 256) {
#if USE_ASYNC_LDS
    // Stage 8 rows x 256 f32 (8 KB) directly global->LDS.
    // 512 x b128 transfers; 2 per thread, tracked by ASYNCcnt.
#pragma unroll
    for (int t = 0; t < 2; ++t) {
      const int idx  = tid + t * 256;      // 0..511
      const int b    = idx >> 6;           // row 0..7
      const int off4 = (idx & 63) * 4;     // f32 offset within row
      const float* g = xc + ((size_t)b * Lc + n) * Dc + d0 + off4;
      __builtin_amdgcn_global_load_async_to_lds_b128(
          (AS1 i32x4*)g, (AS3 i32x4*)&xcs[b][off4], 0, 0);
    }
    __builtin_amdgcn_s_wait_asynccnt(0);
    __syncthreads();
#else
#pragma unroll
    for (int b = 0; b < Bc; ++b)
      xcs[b][tid] = xc[((size_t)b * Lc + n) * Dc + d0 + tid];
    __syncthreads();
#endif

    for (int dd = 0; dd < 256; ++dd) {
      const float e = __expf(Asp[(size_t)(d0 + dd) * Sc + s] * pos);
#pragma unroll
      for (int b = 0; b < Bc; ++b) acc[b] += xcs[b][dd] * e;
    }
    __syncthreads();
  }

#pragma unroll
  for (int b = 0; b < Bc; ++b) {
    const size_t idx = ((size_t)b * Lc + n) * Sc + s;
    yg[idx] = (__bf16)(acc[b] * silu_f(gate[idx]));
  }
}

// ---------------------------------------------------------------------------
// Kernel 5: out = yg @ W_out + b_out + skip(x).  K = S = 1024.
// ---------------------------------------------------------------------------
__global__ __launch_bounds__(256) void mamba_outproj(
    const __bf16* __restrict__ yg, const __bf16* __restrict__ WoutT,
    const float* __restrict__ b_out, const float* __restrict__ x,
    float* __restrict__ out) {
  const int lane = threadIdx.x & 31;
  const int wave = threadIdx.x >> 5;
  const int id   = blockIdx.x * 8 + wave;
  const int NT   = Hc / 64;                // 8
  const int tm   = id / NT;                // 0..127
  const int n0   = (id % NT) * 64;

  const int row    = tm * 16 + (lane & 15);
  const int ncol   = lane & 15;
  const int khalf8 = (lane >> 4) * 8;
  const __bf16* arow = yg + (size_t)row * Sc;
  const __bf16* brow[4];
#pragma unroll
  for (int j = 0; j < 4; ++j)
    brow[j] = WoutT + (size_t)(n0 + j * 16 + ncol) * Sc;

  f32x8 acc[4] = {};
  for (int kb = 0; kb < Sc; kb += 32) {
    __builtin_prefetch(arow + kb + 64, 0, 1);
    bf16x16 a = load_frag16(arow + kb, khalf8);
#pragma unroll
    for (int j = 0; j < 4; ++j) {
      bf16x16 b = load_frag16(brow[j] + kb, khalf8);
      acc[j] = __builtin_amdgcn_wmma_f32_16x16x32_bf16(
          false, a, false, b, (short)0, acc[j], false, false);
    }
  }

  const int mbase = tm * 16 + ((lane >> 4) ? 8 : 0);
#pragma unroll
  for (int j = 0; j < 4; ++j) {
    const int n = n0 + j * 16 + ncol;
    const float bias = b_out[n];
#pragma unroll
    for (int i = 0; i < 8; ++i) {
      const int m = mbase + i;
      out[(size_t)m * Hc + n] = acc[j][i] + bias + x[(size_t)m * Hc + n];
    }
  }
}

// ---------------------------------------------------------------------------
// Host launcher. Workspace layout (MB offsets):
//   xn    bf16 [BL,H]      :  0  (2 MB)
//   WinT  bf16 [2D,H]      :  2  (2 MB)
//   xh    bf16 [BL,D]      :  4  (4 MB)
//   convT bf16 [K,D,D]     :  8  (8 MB)
//   WoutT bf16 [H,S]       : 16  (1 MB)
//   gate  f32  [BL,D]      : 18  (8 MB)
//   xc    f32  [BL,D]      : 26  (8 MB)
//   Asp   f32  [D,S]       : 34  (4 MB)
//   yg    bf16 [BL,S]      : 38  (4 MB)   -> total 42 MB
// ---------------------------------------------------------------------------
extern "C" void kernel_launch(void* const* d_in, const int* in_sizes, int n_in,
                              void* d_out, int out_size, void* d_ws, size_t ws_size,
                              hipStream_t stream) {
  (void)in_sizes; (void)n_in; (void)out_size; (void)ws_size;
  const float* x        = (const float*)d_in[0];
  const float* ln_gamma = (const float*)d_in[1];
  const float* ln_beta  = (const float*)d_in[2];
  const float* W_in     = (const float*)d_in[3];
  const float* b_in     = (const float*)d_in[4];
  const float* conv_w   = (const float*)d_in[5];
  const float* A_param  = (const float*)d_in[6];
  const float* W_out    = (const float*)d_in[7];
  const float* b_out    = (const float*)d_in[8];
  float* out = (float*)d_out;

  char* ws = (char*)d_ws;
  const size_t MB = 1024 * 1024;
  __bf16* xn    = (__bf16*)(ws);
  __bf16* WinT  = (__bf16*)(ws + 2 * MB);
  __bf16* xh    = (__bf16*)(ws + 4 * MB);
  __bf16* convT = (__bf16*)(ws + 8 * MB);
  __bf16* WoutT = (__bf16*)(ws + 16 * MB);
  float*  gate  = (float*)(ws + 18 * MB);
  float*  xc    = (float*)(ws + 26 * MB);
  float*  Asp   = (float*)(ws + 34 * MB);
  __bf16* yg    = (__bf16*)(ws + 38 * MB);

  // 0) One-time weight transposes/converts (independent of activations)
  mamba_transpose_bf16<<<dim3((2 * Dc) / 32, Hc / 32, 1), 256, 0, stream>>>(
      W_in, WinT, Hc, 2 * Dc);                           // [H,2D] -> [2D,H]
  mamba_transpose_bf16<<<dim3(Dc / 32, Dc / 32, 4), 256, 0, stream>>>(
      conv_w, convT, Dc, Dc);                            // [K,D,D] -> [K,D,D]^T
  mamba_transpose_bf16<<<dim3(Hc / 32, Sc / 32, 1), 256, 0, stream>>>(
      W_out, WoutT, Sc, Hc);                             // [S,H] -> [H,S]
  // 4a) softplus(A)
  mamba_softplus<<<(Dc * Sc) / 256, 256, 0, stream>>>(A_param, Asp);

  // 1) LayerNorm -> bf16
  mamba_layernorm<<<BLc, 256, 0, stream>>>(x, ln_gamma, ln_beta, xn);

  // 2) in_proj: (BL/16)*(2D/64) = 4096 wave-strips, 8 waves/block
  mamba_inproj<<<4096 / 8, 256, 0, stream>>>(xn, WinT, b_in, xh, gate);

  // 3) conv + silu: (BL/16)*(D/64) = 2048 wave-strips
  mamba_conv_silu<<<2048 / 8, 256, 0, stream>>>(xh, convT, xc);

  // 4b) SSM einsum + gate: grid (S/256, L)
  mamba_ssm_gate<<<dim3(Sc / 256, Lc), 256, 0, stream>>>(xc, Asp, gate, yg);

  // 5) out_proj + bias + skip: (BL/16)*(H/64) = 1024 wave-strips
  mamba_outproj<<<1024 / 8, 256, 0, stream>>>(yg, WoutT, b_out, x, out);
}
